// WaveletLayer_4561255268860
// MI455X (gfx1250) — matrix operand order repaired and checked by reading, not compiled, not run on hardware
//
#include <hip/hip_runtime.h>
#include <hip/hip_bf16.h>
#include <math.h>

// ---------------------------------------------------------------------------
// WaveletLayer for MI455X (gfx1250, wave32, WMMA).
// direct = x @ lin_w.T  done as bf16 WMMA (f32 accum); wave @ comb_w.T folded
// in as one extra zero-padded K-step; LayerNorm as a second row pass.
// ---------------------------------------------------------------------------

typedef __attribute__((ext_vector_type(16))) __bf16 v16bf;
typedef __attribute__((ext_vector_type(8)))  __bf16 v8bf;
typedef __attribute__((ext_vector_type(4)))  __bf16 v4bf;
typedef __attribute__((ext_vector_type(8)))  float  v8f;

#define B_ROWS  8192
#define IN_DIM  4096
#define OUT_DIM 4096
#define NW      16
#define NH      5
#define ND2     5
#define ND4     6

__constant__ float c_db2[4] = {-0.1294095225512604f, -0.2241438680420134f,
                                0.8365163037378079f, -0.4829629131445341f};
__constant__ float c_db4[8] = {-0.0074578275060427f, -0.0233534968567756f,
                                0.0218081502370024f,  0.1323683939878645f,
                               -0.0197721859913265f, -0.4461007108737368f,
                                0.5055922216746659f, -0.1629171572809054f};

// ---------------- f32 -> bf16 bulk convert (float4 granules) ----------------
__global__ void cvt_bf16_kernel(const float* __restrict__ in,
                                __bf16* __restrict__ out, int n4) {
  int i = blockIdx.x * blockDim.x + threadIdx.x;
  const int stride = gridDim.x * blockDim.x;
  for (; i < n4; i += stride) {
    const float4 f = ((const float4*)in)[i];
    v4bf o;
    o[0] = (__bf16)f.x; o[1] = (__bf16)f.y;
    o[2] = (__bf16)f.z; o[3] = (__bf16)f.w;
    ((v4bf*)out)[i] = o;
  }
}

// ---------------- proj[row, 0..15] = x[row,:] @ proj_w.T + proj_b -----------
__global__ __launch_bounds__(256) void proj_kernel(
    const float* __restrict__ x, const float* __restrict__ pw,
    const float* __restrict__ pb, float* __restrict__ proj) {
  const int row = blockIdx.x;
  const int tid = threadIdx.x;
  const float* xr = x + (size_t)row * IN_DIM;
  float acc[NW];
#pragma unroll
  for (int j = 0; j < NW; ++j) acc[j] = 0.f;
  for (int k = tid; k < IN_DIM; k += 256) {
    const float xv = xr[k];
#pragma unroll
    for (int j = 0; j < NW; ++j) acc[j] = fmaf(xv, pw[j * IN_DIM + k], acc[j]);
  }
#pragma unroll
  for (int j = 0; j < NW; ++j) {
#pragma unroll
    for (int off = 16; off >= 1; off >>= 1) acc[j] += __shfl_xor(acc[j], off, 32);
  }
  __shared__ float red[8][NW];
  const int wv = tid >> 5, lane = tid & 31;
  if (lane == 0) {
#pragma unroll
    for (int j = 0; j < NW; ++j) red[wv][j] = acc[j];
  }
  __syncthreads();
  if (tid < NW) {
    float s = 0.f;
#pragma unroll
    for (int w = 0; w < 8; ++w) s += red[w][tid];
    proj[row * NW + tid] = s + pb[tid];
  }
}

// ---- wave[B,16] in bf16: haar(proj[:, :5]) | broadcast db2/db4 features ----
// NOTE: jnp.roll(filt, shift) only permutes taps; mean(sin(5f)) and sum|f| are
// permutation-invariant, so the shift drops out of the math entirely.
__global__ __launch_bounds__(256) void wave_kernel(
    const float* __restrict__ proj, const float* __restrict__ scales,
    const float* __restrict__ trans, __bf16* __restrict__ waveb) {
  float db[ND2 + ND4];
#pragma unroll
  for (int i = 0; i < ND2; ++i) {
    const int idx = NH + i;
    const float sf = 2.f / (1.f + __expf(-scales[idx]));
    float ssin = 0.f, sabs = 0.f;
#pragma unroll
    for (int t = 0; t < 4; ++t) {
      const float f = sf * c_db2[t];
      ssin += __sinf(5.f * f);
      sabs += fabsf(f);
    }
    const float val = 0.25f * (proj[0 * NW + idx] + proj[1 * NW + idx] +
                               proj[2 * NW + idx] + proj[3 * NW + idx]);
    db[i] = val * (ssin * 0.25f) / (sabs * 0.1f);
  }
#pragma unroll
  for (int i = 0; i < ND4; ++i) {
    const int idx = NH + ND2 + i;
    const float sf = 2.f / (1.f + __expf(-scales[idx]));
    float ssin = 0.f, sabs = 0.f;
#pragma unroll
    for (int t = 0; t < 8; ++t) {
      const float f = sf * c_db4[t];
      ssin += __sinf(5.f * f);
      sabs += fabsf(f);
    }
    const float val = 0.25f * (proj[0 * NW + idx] + proj[1 * NW + idx] +
                               proj[2 * NW + idx] + proj[3 * NW + idx]);
    db[ND2 + i] = val * (ssin * 0.125f) / (sabs * 0.1f);
  }
  int row = blockIdx.x * blockDim.x + threadIdx.x;
  const int stride = gridDim.x * blockDim.x;
  for (; row < B_ROWS; row += stride) {
#pragma unroll
    for (int j = 0; j < NH; ++j) {
      const float sx = (proj[row * NW + j] - trans[j]) / scales[j];
      float h = 0.f;
      if (sx >= 0.f && sx < 0.5f) h = 1.f;
      else if (sx >= 0.5f && sx < 1.f) h = -1.f;
      waveb[row * NW + j] = (__bf16)h;
    }
#pragma unroll
    for (int j = 0; j < ND2 + ND4; ++j)
      waveb[row * NW + NH + j] = (__bf16)db[j];
  }
}

// ---------------- bf16 WMMA GEMM + fused wavelet path + bias ----------------
// A frag (16x32, row-major K): lane<16 -> K {0..7}U{16..23}; lane>=16 -> +8.
// B frag (32x16 from W[n,k] row-major): lane L -> col n=L%16, K same as A frag.
// C/D: n = lane%16, m = j + 8*(lane/16).
__device__ __forceinline__ v16bf load_a_frag(const __bf16* p) {
  const v8bf lo = *(const v8bf*)(p);
  const v8bf hi = *(const v8bf*)(p + 16);
  v16bf r;
#pragma unroll
  for (int i = 0; i < 8; ++i) { r[i] = lo[i]; r[i + 8] = hi[i]; }
  return r;
}

__global__ __launch_bounds__(128, 1) void gemm_wave_kernel(
    const __bf16* __restrict__ xb, const __bf16* __restrict__ wb,
    const __bf16* __restrict__ waveb, const __bf16* __restrict__ combb,
    const float* __restrict__ lin_b, const float* __restrict__ comb_b,
    float* __restrict__ y) {
  const int tid  = threadIdx.x;
  const int wv   = tid >> 5;
  const int lane = tid & 31;
  const int ln   = lane & 15;
  const int hl   = lane >> 4;            // 0: lanes 0-15, 1: lanes 16-31
  const int m0 = blockIdx.y * 128 + (wv >> 1) * 64;   // wave: 64 rows
  const int n0 = blockIdx.x * 128 + (wv & 1) * 64;    // wave: 64 cols
  const int offA = hl * 8;               // A-frag K offset
  const int offB = hl * 8;               // B-frag K offset

  const __bf16* a_ptr[4];
  const __bf16* b_ptr[4];
#pragma unroll
  for (int t = 0; t < 4; ++t) {
    a_ptr[t] = xb + (size_t)(m0 + t * 16 + ln) * IN_DIM + offA;
    b_ptr[t] = wb + (size_t)(n0 + t * 16 + ln) * IN_DIM + offB;
  }

  v8f acc[4][4];
#pragma unroll
  for (int tm = 0; tm < 4; ++tm)
#pragma unroll
    for (int tn = 0; tn < 4; ++tn)
#pragma unroll
      for (int i = 0; i < 8; ++i) acc[tm][tn][i] = 0.f;

  // K-loop unrolled 4x so the rotating prefetch sub-tile index is a
  // compile-time constant (no cndmask select chains in the hot loop).
  for (int k0 = 0; k0 < IN_DIM; k0 += 128) {
    const bool pf = (k0 + 256) < IN_DIM;
#pragma unroll
    for (int u = 0; u < 4; ++u) {
      const int k = k0 + u * 32;
      if (pf) {
        // WGP-scope prefetch, 8 K-steps ahead; u rotates over the 4 M/N
        // sub-tile row streams per lane (all 8 streams covered per 128 K).
        __builtin_prefetch((const void*)(a_ptr[u] + k + 256), 0, 3);
        __builtin_prefetch((const void*)(b_ptr[u] + k + 256), 0, 3);
      }
      v16bf a[4], b[4];
#pragma unroll
      for (int t = 0; t < 4; ++t) a[t] = load_a_frag(a_ptr[t] + k);
#pragma unroll
      for (int t = 0; t < 4; ++t) b[t] = load_a_frag(b_ptr[t] + k);
#pragma unroll
      for (int tm = 0; tm < 4; ++tm)
#pragma unroll
        for (int tn = 0; tn < 4; ++tn)
          acc[tm][tn] = __builtin_amdgcn_wmma_f32_16x16x32_bf16(
              false, a[tm], false, b[tn], (short)0, acc[tm][tn], false, false);
    }
  }

  // Fused wavelet path: one zero-padded K-step (K=16 real, K=16..31 zero).
  {
    v16bf a[4], b[4];
#pragma unroll
    for (int t = 0; t < 4; ++t) {
      const v8bf la = *(const v8bf*)(waveb + (size_t)(m0 + t * 16 + ln) * NW + offA);
#pragma unroll
      for (int i = 0; i < 8; ++i) { a[t][i] = la[i]; a[t][i + 8] = (__bf16)0.f; }
      const v8bf lb = *(const v8bf*)(combb + (size_t)(n0 + t * 16 + ln) * NW + offB);
#pragma unroll
      for (int i = 0; i < 8; ++i) { b[t][i] = lb[i]; b[t][i + 8] = (__bf16)0.f; }
    }
#pragma unroll
    for (int tm = 0; tm < 4; ++tm)
#pragma unroll
      for (int tn = 0; tn < 4; ++tn)
        acc[tm][tn] = __builtin_amdgcn_wmma_f32_16x16x32_bf16(
            false, a[tm], false, b[tn], (short)0, acc[tm][tn], false, false);
  }

  // Epilogue: + lin_b + comb_b, store f32 y.
#pragma unroll
  for (int tn = 0; tn < 4; ++tn) {
    const int n = n0 + tn * 16 + ln;
    const float bias = lin_b[n] + comb_b[n];
#pragma unroll
    for (int tm = 0; tm < 4; ++tm) {
      const int mb = m0 + tm * 16 + hl * 8;
#pragma unroll
      for (int j = 0; j < 8; ++j)
        y[(size_t)(mb + j) * OUT_DIM + n] = acc[tm][tn][j] + bias;
    }
  }
}

// ---------------- row-wise LayerNorm (values held in registers) -------------
__global__ __launch_bounds__(256) void ln_kernel(float* __restrict__ y,
                                                 const float* __restrict__ g,
                                                 const float* __restrict__ bt) {
  const int row = blockIdx.x, tid = threadIdx.x;
  float* yr = y + (size_t)row * OUT_DIM;
  float v[16], s = 0.f, s2 = 0.f;
#pragma unroll
  for (int i = 0; i < 16; ++i) {
    const float t = yr[tid + i * 256];
    v[i] = t; s += t; s2 += t * t;
  }
#pragma unroll
  for (int off = 16; off >= 1; off >>= 1) {
    s  += __shfl_xor(s, off, 32);
    s2 += __shfl_xor(s2, off, 32);
  }
  __shared__ float sh[2][8];
  if ((tid & 31) == 0) { sh[0][tid >> 5] = s; sh[1][tid >> 5] = s2; }
  __syncthreads();
  float ts = 0.f, ts2 = 0.f;
#pragma unroll
  for (int w = 0; w < 8; ++w) { ts += sh[0][w]; ts2 += sh[1][w]; }
  const float mu  = ts * (1.f / OUT_DIM);
  const float var = ts2 * (1.f / OUT_DIM) - mu * mu;
  const float inv = rsqrtf(var + 1e-5f);
#pragma unroll
  for (int i = 0; i < 16; ++i) {
    const int k = tid + i * 256;
    yr[k] = (v[i] - mu) * inv * g[k] + bt[k];
  }
}

// ---------------------------------------------------------------------------
extern "C" void kernel_launch(void* const* d_in, const int* in_sizes, int n_in,
                              void* d_out, int out_size, void* d_ws, size_t ws_size,
                              hipStream_t stream) {
  (void)in_sizes; (void)n_in; (void)out_size; (void)ws_size;
  const float* x      = (const float*)d_in[0];
  const float* lin_w  = (const float*)d_in[1];
  const float* lin_b  = (const float*)d_in[2];
  const float* proj_w = (const float*)d_in[3];
  const float* proj_b = (const float*)d_in[4];
  const float* scales = (const float*)d_in[5];
  const float* trans  = (const float*)d_in[6];
  const float* comb_w = (const float*)d_in[7];
  const float* comb_b = (const float*)d_in[8];
  const float* ln_g   = (const float*)d_in[9];
  const float* ln_bt  = (const float*)d_in[10];
  float* y = (float*)d_out;

  // Workspace layout (~97 MiB total; xb+wb are L2-resident for the GEMM):
  char* ws = (char*)d_ws;
  size_t off = 0;
  __bf16* xb    = (__bf16*)(ws + off); off += (size_t)B_ROWS * IN_DIM * 2;   // 64 MiB
  __bf16* wb    = (__bf16*)(ws + off); off += (size_t)OUT_DIM * IN_DIM * 2;  // 32 MiB
  __bf16* combb = (__bf16*)(ws + off); off += (size_t)OUT_DIM * NW * 2;      // 128 KiB
  float*  proj  = (float*) (ws + off); off += (size_t)B_ROWS * NW * 4;       // 512 KiB
  __bf16* waveb = (__bf16*)(ws + off);                                       // 256 KiB

  cvt_bf16_kernel<<<4096, 256, 0, stream>>>(x, xb, B_ROWS * IN_DIM / 4);
  cvt_bf16_kernel<<<4096, 256, 0, stream>>>(lin_w, wb, OUT_DIM * IN_DIM / 4);
  cvt_bf16_kernel<<<64, 256, 0, stream>>>(comb_w, combb, OUT_DIM * NW / 4);
  proj_kernel<<<B_ROWS, 256, 0, stream>>>(x, proj_w, proj_b, proj);
  wave_kernel<<<32, 256, 0, stream>>>(proj, scales, trans, waveb);

  dim3 grid(OUT_DIM / 128, B_ROWS / 128);   // 32 x 64 workgroups, 4 waves each
  gemm_wave_kernel<<<grid, 128, 0, stream>>>(xb, wb, waveb, combb,
                                             lin_b, comb_b, y);
  ln_kernel<<<B_ROWS, 256, 0, stream>>>(y, ln_g, ln_bt);
}
